// MyLSTM_15590731284701
// MI455X (gfx1250) — compile-verified, matrix operand order checked by reading, not run on hardware
//
#include <hip/hip_runtime.h>
#include <hip/hip_bf16.h>

// ---------------------------------------------------------------------------
// LSTM on MI455X (gfx1250): fused per-timestep kernel using
// V_WMMA_F32_16X16X32_BF16 (bf16 operands / fp32 accumulate), with
// double-buffered LDS fed by GLOBAL_LOAD_ASYNC_TO_LDS_B128.
// ---------------------------------------------------------------------------

typedef __bf16 bf16_t;
typedef __attribute__((ext_vector_type(16))) __bf16 v16bf;
typedef __attribute__((ext_vector_type(8)))  float  v8f;

typedef int v4i_t __attribute__((ext_vector_type(4)));
typedef __attribute__((address_space(1))) v4i_t g_v4i;  // global int4
typedef __attribute__((address_space(3))) v4i_t l_v4i;  // LDS int4

#define T_STEPS 128
#define BATCH   1024
#define EDIM    1024
#define HDIM    1024

// --- fast activations (TRANS ops co-execute with WMMA on CDNA5) -------------
__device__ __forceinline__ float sigf(float x) {
  return 1.0f / (1.0f + __expf(-x));
}
__device__ __forceinline__ float tanhfast(float x) {
  float e = __expf(-2.0f * x);
  return (1.0f - e) / (1.0f + e);
}

// 16-byte global -> LDS copy. Prefer the CDNA5 async-to-LDS path (ASYNCcnt),
// fall back to a synchronous VGPR-staged copy.
__device__ __forceinline__ void cp16_g2l(const bf16_t* g, bf16_t* l) {
#if defined(__has_builtin) && __has_builtin(__builtin_amdgcn_global_load_async_to_lds_b128)
  __builtin_amdgcn_global_load_async_to_lds_b128(
      (g_v4i*)(uintptr_t)g, (l_v4i*)(unsigned)(uintptr_t)l, 0, 0);
#else
  *(uint4*)l = *(const uint4*)g;
#endif
}
__device__ __forceinline__ void wait_g2l() {
#if defined(__has_builtin) && __has_builtin(__builtin_amdgcn_s_wait_asynccnt)
  __builtin_amdgcn_s_wait_asynccnt(0);
#else
  asm volatile("s_wait_asynccnt 0" ::: "memory");
#endif
}

// Fragment-layout offset helper: a 64(d) x 32(k) bf16 tile is stored as 4
// sub-tiles, each in exact WMMA v16bf per-lane order:
//   lane = (d%16) + 16*((k%16)/8),  slot = (k/16)*8 + (k%8)
// so each lane's operand is one contiguous 32B LDS read, and every aligned
// 8-element (16B) k-chunk of a row maps to one aligned 16B LDS chunk.
__device__ __forceinline__ int frag_slot_off(int d, int k8) {
  int sub  = d >> 4;
  int lane = (d & 15) + (((k8 >> 3) & 1) << 4);
  int slot = (k8 >> 4) << 3;
  return ((sub << 5) + lane) * 16 + slot;
}

// --- prep kernels ------------------------------------------------------------
__global__ void f32_to_bf16_kernel(const float* __restrict__ src,
                                   bf16_t* __restrict__ dst, long n) {
  long i = (long)blockIdx.x * blockDim.x + threadIdx.x;
  long stride = (long)gridDim.x * blockDim.x;
  for (; i < n; i += stride) dst[i] = (bf16_t)src[i];
}

// src[rows][cols] fp32 -> dst[cols][rows] bf16 (LDS-tiled transpose)
__global__ void transpose_to_bf16_kernel(const float* __restrict__ src,
                                         bf16_t* __restrict__ dst,
                                         int rows, int cols) {
  __shared__ float tile[32][33];
  int r0 = blockIdx.y * 32, c0 = blockIdx.x * 32;
  int tx = threadIdx.x & 31, ty = threadIdx.x >> 5;  // 256 thr: ty = 0..7
  #pragma unroll
  for (int i = ty; i < 32; i += 8)
    tile[i][tx] = src[(long)(r0 + i) * cols + c0 + tx];
  __syncthreads();
  #pragma unroll
  for (int i = ty; i < 32; i += 8)
    dst[(long)(c0 + i) * rows + r0 + tx] = (bf16_t)tile[tx][i];
}

__global__ void init_state_kernel(bf16_t* __restrict__ hT,
                                  float* __restrict__ c, long n) {
  long i = (long)blockIdx.x * blockDim.x + threadIdx.x;
  long stride = (long)gridDim.x * blockDim.x;
  for (; i < n; i += stride) { hT[i] = (bf16_t)0.0f; c[i] = 0.0f; }
}

// --- fused LSTM step ---------------------------------------------------------
// Block: 256 threads (8 waves), 64x64 output tile for ALL 4 gates, then the
// cell update in-register.
//   pre_g[m][n] = sum_e Xt[m][e]*Wi_g[e][n] + sum_k Wh_g[m][k]*h[k][n] + b_g[n]
// LDS regions (per buffer): 0..3 = A per gate (region 0 doubles as shared A),
//                           4..7 = B per gate (region 4 doubles as shared B).
__global__ __launch_bounds__(256)
void lstm_step_kernel(const bf16_t* __restrict__ Xt,     // [BATCH][EDIM]
                      const bf16_t* __restrict__ WiT,    // [4][HDIM][EDIM] (n-major)
                      const bf16_t* __restrict__ Wh,     // [4][HDIM][HDIM] row-major
                      const float* __restrict__ bi,
                      const float* __restrict__ bf_,
                      const float* __restrict__ bg,
                      const float* __restrict__ bo,
                      const bf16_t* __restrict__ hprevT, // [HDIM(n)][HDIM(k)]
                      bf16_t* __restrict__ hnextT,       // [HDIM(n)][HDIM(k)]
                      float* __restrict__ c_state,       // [BATCH][HDIM] fp32
                      float* __restrict__ h_out) {       // [BATCH][HDIM] fp32
  __shared__ __attribute__((aligned(32))) bf16_t lds[2][8 * 2048];  // 2 x 32KB

  const int tid  = threadIdx.x;
  const int lane = tid & 31;
  const int wave = tid >> 5;
  const int mi   = wave & 3;          // sub-tile row (0..3)
  const int nj   = (wave >> 2) << 1;  // sub-tile col pair base (0 or 2)
  const int mTile = blockIdx.y << 6;
  const int nTile = blockIdx.x << 6;

  v8f acc[4][2] = {};  // [gate][nj sub-tile]

  // staging decomposition: each thread owns one 16B chunk per region
  const int cd  = tid >> 2;           // 0..63
  const int ck8 = (tid & 3) << 3;     // 0,8,16,24
  const int F   = frag_slot_off(cd, ck8);

  const bf16_t* xRow = Xt     + (size_t)(mTile + cd) * EDIM + ck8;
  const bf16_t* hRow = hprevT + (size_t)(nTile + cd) * HDIM + ck8;
  const bf16_t* wiRow[4];
  const bf16_t* whRow[4];
  #pragma unroll
  for (int g = 0; g < 4; ++g) {
    wiRow[g] = WiT + ((size_t)g * HDIM + nTile + cd) * EDIM + ck8;
    whRow[g] = Wh  + ((size_t)g * HDIM + mTile + cd) * HDIM + ck8;
  }

  auto fill = [&](int ks, bf16_t* buf) {
    const int k0 = (ks & 31) << 5;
    if (ks < 32) {             // input phase: A shared (X), B per gate (WiT)
      cp16_g2l(xRow + k0, buf + 0 * 2048 + F);
      #pragma unroll
      for (int g = 0; g < 4; ++g)
        cp16_g2l(wiRow[g] + k0, buf + (4 + g) * 2048 + F);
    } else {                   // recurrent phase: A per gate (Wh), B shared (hT)
      cp16_g2l(hRow + k0, buf + 4 * 2048 + F);
      #pragma unroll
      for (int g = 0; g < 4; ++g)
        cp16_g2l(whRow[g] + k0, buf + g * 2048 + F);
    }
  };

  const int aLane  = (mi * 32 + lane) * 16;
  const int bLane0 = ((nj)     * 32 + lane) * 16;
  const int bLane1 = ((nj + 1) * 32 + lane) * 16;

  fill(0, lds[0]);

  // ---- phase 1: K over EDIM (A = Xt shared, B = WiT per gate) ----
  for (int ks = 0; ks < 32; ++ks) {
    bf16_t* cur = lds[ks & 1];
    wait_g2l();
    __syncthreads();
    fill(ks + 1, lds[(ks + 1) & 1]);
    v16bf a = *(const v16bf*)&cur[0 * 2048 + aLane];
    #pragma unroll
    for (int g = 0; g < 4; ++g) {
      v16bf b0 = *(const v16bf*)&cur[(4 + g) * 2048 + bLane0];
      v16bf b1 = *(const v16bf*)&cur[(4 + g) * 2048 + bLane1];
      acc[g][0] = __builtin_amdgcn_wmma_f32_16x16x32_bf16(
          false, a, false, b0, (short)0, acc[g][0], false, false);
      acc[g][1] = __builtin_amdgcn_wmma_f32_16x16x32_bf16(
          false, a, false, b1, (short)0, acc[g][1], false, false);
    }
  }

  // ---- phase 2: K over HDIM (A = Wh per gate, B = h^T shared) ----
  for (int ks = 32; ks < 64; ++ks) {
    bf16_t* cur = lds[ks & 1];
    wait_g2l();
    __syncthreads();
    if (ks + 1 < 64) fill(ks + 1, lds[(ks + 1) & 1]);
    v16bf b0 = *(const v16bf*)&cur[4 * 2048 + bLane0];
    v16bf b1 = *(const v16bf*)&cur[4 * 2048 + bLane1];
    #pragma unroll
    for (int g = 0; g < 4; ++g) {
      v16bf a = *(const v16bf*)&cur[g * 2048 + aLane];
      acc[g][0] = __builtin_amdgcn_wmma_f32_16x16x32_bf16(
          false, a, false, b0, (short)0, acc[g][0], false, false);
      acc[g][1] = __builtin_amdgcn_wmma_f32_16x16x32_bf16(
          false, a, false, b1, (short)0, acc[g][1], false, false);
    }
  }

  // ---- cell update epilogue (D layout: m = r + 8*(lane/16), n = lane%16) ----
  const float* bp[4] = {bi, bf_, bg, bo};
  #pragma unroll
  for (int j = 0; j < 2; ++j) {
    const int n = nTile + ((nj + j) << 4) + (lane & 15);
    float bv[4];
    #pragma unroll
    for (int g = 0; g < 4; ++g) bv[g] = bp[g][n];
    const int mBase = mTile + (mi << 4) + ((lane >> 4) << 3);
    #pragma unroll
    for (int r = 0; r < 8; ++r) {
      const int m = mBase + r;
      const size_t idx = (size_t)m * HDIM + n;
      float pi = acc[0][j][r] + bv[0];
      float pf = acc[1][j][r] + bv[1];
      float pg = acc[2][j][r] + bv[2];
      float po = acc[3][j][r] + bv[3];
      float ig = sigf(pi), fg = sigf(pf), gv = tanhfast(pg), og = sigf(po);
      float cn = fg * c_state[idx] + ig * gv;
      c_state[idx] = cn;
      float hn = og * tanhfast(cn);
      h_out[idx] = hn;
      hnextT[(size_t)n * HDIM + m] = (bf16_t)hn;  // transposed for next step's B
    }
  }
}

// ---------------------------------------------------------------------------
extern "C" void kernel_launch(void* const* d_in, const int* in_sizes, int n_in,
                              void* d_out, int out_size, void* d_ws, size_t ws_size,
                              hipStream_t stream) {
  const float* X   = (const float*)d_in[0];
  const float* Wii = (const float*)d_in[1];
  const float* Whi = (const float*)d_in[2];
  const float* bi  = (const float*)d_in[3];
  const float* Wif = (const float*)d_in[4];
  const float* Whf = (const float*)d_in[5];
  const float* bf_ = (const float*)d_in[6];
  const float* Wig = (const float*)d_in[7];
  const float* Whg = (const float*)d_in[8];
  const float* bg  = (const float*)d_in[9];
  const float* Wio = (const float*)d_in[10];
  const float* Who = (const float*)d_in[11];
  const float* bo  = (const float*)d_in[12];
  float* out = (float*)d_out;

  char* ws = (char*)d_ws;
  size_t off = 0;
  bf16_t* Xbf  = (bf16_t*)(ws + off); off += (size_t)T_STEPS * BATCH * EDIM * 2;
  bf16_t* WiT  = (bf16_t*)(ws + off); off += 4ull * HDIM * EDIM * 2;
  bf16_t* Whbf = (bf16_t*)(ws + off); off += 4ull * HDIM * HDIM * 2;
  bf16_t* hT0  = (bf16_t*)(ws + off); off += (size_t)HDIM * HDIM * 2;
  bf16_t* hT1  = (bf16_t*)(ws + off); off += (size_t)HDIM * HDIM * 2;
  float*  cst  = (float*) (ws + off); off += (size_t)BATCH * HDIM * 4;

  // --- per-call prep: bf16 conversion / transposes / state init ---
  f32_to_bf16_kernel<<<4096, 256, 0, stream>>>(
      X, Xbf, (long)T_STEPS * BATCH * EDIM);

  const float* whs[4] = {Whi, Whf, Whg, Who};
  for (int g = 0; g < 4; ++g)
    f32_to_bf16_kernel<<<1024, 256, 0, stream>>>(
        whs[g], Whbf + (size_t)g * HDIM * HDIM, (long)HDIM * HDIM);

  const float* wis[4] = {Wii, Wif, Wig, Wio};
  dim3 tgrid(HDIM / 32, EDIM / 32);
  for (int g = 0; g < 4; ++g)
    transpose_to_bf16_kernel<<<tgrid, 256, 0, stream>>>(
        wis[g], WiT + (size_t)g * HDIM * EDIM, EDIM, HDIM);

  init_state_kernel<<<1024, 256, 0, stream>>>(hT0, cst, (long)BATCH * HDIM);

  // --- sequential recurrence: one fused kernel per timestep ---
  dim3 sgrid(HDIM / 64, BATCH / 64);  // (16,16) -> 256 workgroups
  for (int t = 0; t < T_STEPS; ++t) {
    const bf16_t* hp = (t & 1) ? hT1 : hT0;
    bf16_t*       hn = (t & 1) ? hT0 : hT1;
    lstm_step_kernel<<<sgrid, 256, 0, stream>>>(
        Xbf + (size_t)t * BATCH * EDIM, WiT, Whbf,
        bi, bf_, bg, bo, hp, hn, cst,
        out + (size_t)t * BATCH * HDIM);
  }

  // --- tail outputs: h_T and c_T after hidden_sequence ---
  (void)hipMemcpyAsync(out + (size_t)T_STEPS * BATCH * HDIM,
                       out + (size_t)(T_STEPS - 1) * BATCH * HDIM,
                       (size_t)BATCH * HDIM * sizeof(float),
                       hipMemcpyDeviceToDevice, stream);
  (void)hipMemcpyAsync(out + (size_t)T_STEPS * BATCH * HDIM + (size_t)BATCH * HDIM,
                       cst,
                       (size_t)BATCH * HDIM * sizeof(float),
                       hipMemcpyDeviceToDevice, stream);
}
